// DA_TRANS_lang_53145925321004
// MI455X (gfx1250) — matrix operand order repaired and checked by reading, not compiled, not run on hardware
//
#include <hip/hip_runtime.h>
#include <cstddef>

// ---------------------------------------------------------------------------
// CDNA5 (gfx1250, wave32) domain-adapted biLSTM LM.
// All large GEMMs: v_wmma_f32_16x16x32_bf16, operands pre-swizzled into the
// ISA per-lane fragment order so the hot loops are pure b128 loads + WMMA.
// ---------------------------------------------------------------------------

typedef __attribute__((ext_vector_type(16))) __bf16        v16bf;
typedef __attribute__((ext_vector_type(8)))  float         v8f;
typedef __attribute__((ext_vector_type(4)))  unsigned int  u32x4;

union FragBf {
    v16bf          v;
    unsigned short s[16];
    unsigned int   u[8];
    u32x4          q[2];
};

__device__ __forceinline__ unsigned short f2bf(float f) {
    unsigned int u = __float_as_uint(f);
    u += 0x7FFFu + ((u >> 16) & 1u);          // round to nearest even
    return (unsigned short)(u >> 16);
}

__device__ __forceinline__ float sigm(float x) { return 1.0f / (1.0f + __expf(-x)); }

// --- fragment-order ("swizzled") element offsets --------------------------
// Tiles of 512 bf16 (one wave-operand). Tile (t, kk) at ((t*KT + kk) * 512),
// lane L owns 16 contiguous elements at lane*16.
// A (16x32, MxK): value (m, k): lane = 16*((k>>3)&1) + m;
//   elem = interleaved {2v | 2v+8} K pattern of ISA 7.12.2.
__device__ __forceinline__ long aswz(int m, int k, int KT) {
    const int mt   = m >> 4, ml = m & 15;
    const int kk   = k >> 5, kp = k & 31;
    const int half = (kp >> 3) & 1;
    const int base = kp & 7;
    const int v    = (kp < 16) ? (base >> 1) : 4 + (base >> 1);
    const int elem = 2 * v + (base & 1);
    return (((long)mt * KT + kk) << 9) + (((half << 4) + ml) << 4) + elem;
}
// B (32x16, KxN): value (k, n): lane = 16*((k>>3)&1) + (n&15); elem = (k&7) + 8*((k>>4)&1).
__device__ __forceinline__ long bswz(int k, int n, int KT) {
    const int nt   = n >> 4, nl = n & 15;
    const int kk   = k >> 5;
    const int half = (k >> 3) & 1;
    const int e    = (k & 7) + (((k >> 4) & 1) << 3);
    return (((long)nt * KT + kk) << 9) + (((half << 4) + nl) << 4) + e;
}

// ---------------------------------------------------------------------------
// Generic batched bf16 WMMA GEMM on swizzled operands.
// Block = 128 threads = 4 waves; wave w -> output tile (blockIdx.y, bx*4+w).
// Inner loop: 2x b128 (A) + 2x b128 (B) + 1x v_wmma per 16x16x32.
// EPI: 0 = none, 1 = leaky_relu(0.01)
// ---------------------------------------------------------------------------
template <int EPI>
__global__ __launch_bounds__(128) void wmma_gemm(
    const unsigned short* __restrict__ A, long aBatch,
    const unsigned short* __restrict__ Bm, long bBatch,
    const float* __restrict__ bias, long biasBatch,
    float* __restrict__ C, int ldc, long cBatch,
    int N, int KT)
{
    const long z = blockIdx.z;
    A  += z * aBatch;
    Bm += z * bBatch;
    C  += z * cBatch;
    if (bias) bias += z * biasBatch;

    const int wave = threadIdx.x >> 5;
    const int lane = threadIdx.x & 31;
    const int half = lane >> 4;
    const int nl   = lane & 15;
    const int mt   = blockIdx.y;
    const int nt   = blockIdx.x * 4 + wave;

    const unsigned short* Ap = A  + (((long)mt * KT) << 9) + (lane << 4);
    const unsigned short* Bp = Bm + (((long)nt * KT) << 9) + (lane << 4);

    v8f acc = {};
    for (int kk = 0; kk < KT; ++kk) {
        FragBf af, bfr;
        af.q[0]  = *(const u32x4*)(Ap);
        af.q[1]  = *(const u32x4*)(Ap + 8);
        bfr.q[0] = *(const u32x4*)(Bp);
        bfr.q[1] = *(const u32x4*)(Bp + 8);
        if (kk + 1 < KT) __builtin_prefetch(Bp + 512, 0, 1);
        acc = __builtin_amdgcn_wmma_f32_16x16x32_bf16(
            false, af.v, false, bfr.v, (short)0, acc, false, false);
        Ap += 512;
        Bp += 512;
    }

    const int cn = nt * 16 + nl;
    if (cn < N) {
        const float bv = bias ? bias[cn] : 0.0f;
#pragma unroll
        for (int r = 0; r < 8; ++r) {
            const int cm = mt * 16 + r + half * 8;
            float v = acc[r] + bv;
            if (EPI == 1) v = v > 0.0f ? v : 0.01f * v;
            C[(long)cm * ldc + cn] = v;
        }
    }
}

// ---------------------------------------------------------------------------
// doms[16,300] @ W.T + bias -> out[16, N].  One thread per output column n;
// 16 batch accumulators so each weight row is streamed exactly once.
// ---------------------------------------------------------------------------
__global__ __launch_bounds__(256) void dom_linear(
    const float* __restrict__ doms, const float* __restrict__ W,
    const float* __restrict__ bias, float* __restrict__ out, int N)
{
    __shared__ float dsh[16 * 300];
    for (int i = threadIdx.x; i < 16 * 300; i += 256) dsh[i] = doms[i];
    __syncthreads();

    const int n = blockIdx.x * 256 + threadIdx.x;
    if (n >= N) return;

    float acc[16];
#pragma unroll
    for (int b = 0; b < 16; ++b) acc[b] = 0.0f;

    const float* wr = W + (long)n * 300;
    for (int k = 0; k < 300; ++k) {
        const float w = wr[k];
#pragma unroll
        for (int b = 0; b < 16; ++b) acc[b] += dsh[b * 300 + k] * w;
    }
    const float bb = bias[n];
#pragma unroll
    for (int b = 0; b < 16; ++b) out[(long)b * N + n] = acc[b] + bb;
}

// --- operand preparation: fp32 -> swizzled, padded bf16 -------------------
// A-operand from row-major [sRows, sCols] (batched)
__global__ void swizzleA_f32(const float* __restrict__ src, long sBatch, int sRows, int sCols,
                             unsigned short* __restrict__ dst, long dBatch, int Mp, int Kp)
{
    const long z = blockIdx.z;
    const long i = (long)blockIdx.x * blockDim.x + threadIdx.x;
    if (i >= (long)Mp * Kp) return;
    const int m = (int)(i / Kp);
    const int k = (int)(i % Kp);
    unsigned short v = 0;
    if (m < sRows && k < sCols) v = f2bf(src[z * sBatch + (long)m * sCols + k]);
    dst[z * dBatch + aswz(m, k, Kp >> 5)] = v;
}

// B-operand from row-major [K, N] (batched)
__global__ void swizzleB_f32(const float* __restrict__ src, long sBatch, int sRows, int sCols,
                             unsigned short* __restrict__ dst, long dBatch, int Kp, int Np)
{
    const long z = blockIdx.z;
    const long i = (long)blockIdx.x * blockDim.x + threadIdx.x;
    if (i >= (long)Kp * Np) return;
    const int k = (int)(i / Np);
    const int n = (int)(i % Np);
    unsigned short v = 0;
    if (k < sRows && n < sCols) v = f2bf(src[z * sBatch + (long)k * sCols + n]);
    dst[z * dBatch + bswz(k, n, Kp >> 5)] = v;
}

// B-operand from torch Linear weight W[Out, In]: B[k][n] = W[n][k]
__global__ void swizzleB_T(const float* __restrict__ W, int Out, int In,
                           unsigned short* __restrict__ dst, int Kp, int Np)
{
    const long i = (long)blockIdx.x * blockDim.x + threadIdx.x;
    if (i >= (long)Kp * Np) return;
    const int k = (int)(i / Np);
    const int n = (int)(i % Np);
    unsigned short v = 0;
    if (k < In && n < Out) v = f2bf(W[(long)n * In + k]);
    dst[bswz(k, n, Kp >> 5)] = v;
}

// A-operand column block (for concat): dst rows x Kp, src [rows, srcCols]
__global__ void swizzleA_cols(const float* __restrict__ src, int srcCols,
                              unsigned short* __restrict__ dst, int Kp, int colOff, long total)
{
    const long i = (long)blockIdx.x * blockDim.x + threadIdx.x;
    if (i >= total) return;
    const int  r = (int)(i / srcCols);
    const int  c = (int)(i % srcCols);
    dst[aswz(r, colOff + c, Kp >> 5)] = f2bf(src[i]);
}

// ---------------------------------------------------------------------------
// Attention finalize + blend (stage 1): aw = sigmoid(h.m2 + b2);
// weighted (A-operand for the gate GEMMs) written swizzled, Kp = 320.
// One block per (b,s), 320 threads.
// ---------------------------------------------------------------------------
__global__ __launch_bounds__(320) void attn_blend1(
    const float* __restrict__ hattn, const float* __restrict__ attn_t,
    const float* __restrict__ inputs, const float* __restrict__ trans_in,
    unsigned short* __restrict__ wbf)
{
    const int b = blockIdx.x >> 8;
    const int s = blockIdx.x & 255;
    const long r = (long)(b * 256 + s);
    const float* at = attn_t + (long)b * 19329;

    __shared__ float red[64];
    __shared__ float aw_sh;
    const int tid = threadIdx.x;
    if (tid < 64) red[tid] = hattn[r * 64 + tid] * at[19264 + tid];
    __syncthreads();
    if (tid == 0) {
        float sum = at[19328];
        for (int j = 0; j < 64; ++j) sum += red[j];
        aw_sh = sigm(sum);
    }
    __syncthreads();
    const float aw = aw_sh;

    unsigned short v = 0;
    if (tid < 300) {
        const float x = inputs[r * 300 + tid] * aw + trans_in[r * 300 + tid] * (1.0f - aw);
        v = f2bf(x);
    }
    wbf[aswz((int)r, tid, 10)] = v;       // Kp=320 -> KT=10
}

// Stage 2 (OUT_D = 100, Kp = 128). One block per (b,s), 128 threads.
__global__ __launch_bounds__(128) void attn_blend2(
    const float* __restrict__ h2, const float* __restrict__ attn_to,
    const float* __restrict__ proc, const float* __restrict__ trans_proc,
    unsigned short* __restrict__ pabf)
{
    const int b = blockIdx.x >> 8;
    const int s = blockIdx.x & 255;
    const long r = (long)(b * 256 + s);
    const float* at = attn_to + (long)b * 6529;

    __shared__ float red[64];
    __shared__ float aw_sh;
    const int tid = threadIdx.x;
    if (tid < 64) red[tid] = h2[r * 64 + tid] * at[6464 + tid];
    __syncthreads();
    if (tid == 0) {
        float sum = at[6528];
        for (int j = 0; j < 64; ++j) sum += red[j];
        aw_sh = sigm(sum);
    }
    __syncthreads();
    const float aw = aw_sh;

    unsigned short v = 0;
    if (tid < 100) {
        const float x = proc[r * 100 + tid] * aw + trans_proc[r * 100 + tid] * (1.0f - aw);
        v = f2bf(x);
    }
    pabf[aswz((int)r, tid, 4)] = v;       // Kp=128 -> KT=4
}

// ---------------------------------------------------------------------------
// Persistent biLSTM: grid.x = 2 (fwd/bwd), 1024 threads = 32 waves per block.
// h lives in LDS in A-fragment order (16 tiles of 512 bf16); c fp32 in LDS.
// Wave w owns hidden cols w*16..w*16+15; its 4 gate tiles sit at
// N = gate*512 + w*16 so i/f/g/o stay in that wave's accumulators.
// Recurrent matmul: 2x ds_load_b128 (A) + 2x global_load_b128 (B) per WMMA.
// ---------------------------------------------------------------------------
__global__ __launch_bounds__(1024) void lstm_kernel(
    const float* __restrict__ pre_f, const float* __restrict__ pre_b,
    const unsigned short* __restrict__ WhhT_f, const unsigned short* __restrict__ WhhT_b,
    const float* __restrict__ bias_f, const float* __restrict__ bias_b,
    float* __restrict__ hout_f, float* __restrict__ hout_b)
{
    const int dir = blockIdx.x;
    const float*          pre  = dir ? pre_b  : pre_f;
    const unsigned short* WhhT = dir ? WhhT_b : WhhT_f;
    const float*          bias = dir ? bias_b : bias_f;
    float*                hout = dir ? hout_b : hout_f;

    __shared__ __align__(16) unsigned short h_sw[16 * 512];   // 16 KB, A-fragment order
    __shared__ float                         c_st[16 * 512];  // 32 KB

    for (int i = threadIdx.x; i < 16 * 512; i += 1024) { h_sw[i] = 0; c_st[i] = 0.0f; }
    __syncthreads();

    const int wave = threadIdx.x >> 5;
    const int lane = threadIdx.x & 31;
    const int half = lane >> 4;
    const int nl   = lane & 15;
    const int col  = wave * 16 + nl;          // hidden index owned by this lane
    const float bi  = bias[0 * 512 + col];
    const float bfv = bias[1 * 512 + col];
    const float bg  = bias[2 * 512 + col];
    const float bo  = bias[3 * 512 + col];

    // precomputed swizzled-LDS write coordinates for k = col
    const int kk0   = col >> 5;
    const int kp0   = col & 31;
    const int halfA = (kp0 >> 3) & 1;
    const int baseA = kp0 & 7;
    const int vA    = (kp0 < 16) ? (baseA >> 1) : 4 + (baseA >> 1);
    const int elemA = 2 * vA + (baseA & 1);

    for (int step = 0; step < 256; ++step) {
        const int t = dir ? (255 - step) : step;

        v8f acc[4] = {};                      // gate tiles i,f,g,o
        for (int kk = 0; kk < 16; ++kk) {
            FragBf af;
            const unsigned short* hp = &h_sw[(kk << 9) + (lane << 4)];
            af.q[0] = *(const u32x4*)(hp);
            af.q[1] = *(const u32x4*)(hp + 8);
#pragma unroll
            for (int g = 0; g < 4; ++g) {
                const int nt = g * 32 + wave;
                const unsigned short* bp = WhhT + (((long)nt * 16 + kk) << 9) + (lane << 4);
                FragBf bfr;
                bfr.q[0] = *(const u32x4*)(bp);
                bfr.q[1] = *(const u32x4*)(bp + 8);
                acc[g] = __builtin_amdgcn_wmma_f32_16x16x32_bf16(
                    false, af.v, false, bfr.v, (short)0, acc[g], false, false);
            }
        }
        __syncthreads();                      // all waves done reading h_sw

#pragma unroll
        for (int r = 0; r < 8; ++r) {
            const int m = r + half * 8;       // batch index
            const long rowp = (long)(m * 256 + t) * 2048;
            const float gi = acc[0][r] + pre[rowp + 0 * 512 + col] + bi;
            const float gf = acc[1][r] + pre[rowp + 1 * 512 + col] + bfv;
            const float gg = acc[2][r] + pre[rowp + 2 * 512 + col] + bg;
            const float go = acc[3][r] + pre[rowp + 3 * 512 + col] + bo;
            const float c_old = c_st[m * 512 + col];
            const float c_new = sigm(gf) * c_old + sigm(gi) * tanhf(gg);
            const float h     = sigm(go) * tanhf(c_new);
            c_st[m * 512 + col] = c_new;
            h_sw[(kk0 << 9) + (((halfA << 4) + m) << 4) + elemA] = f2bf(h);
            hout[(long)(m * 256 + t) * 512 + col] = h;
        }
        __syncthreads();                      // h_sw updated before next step
    }
}

__global__ void tgt_copy(const int* __restrict__ t, int* __restrict__ dst, int n)
{
    const int i = blockIdx.x * blockDim.x + threadIdx.x;
    if (i < n) dst[i] = t[i];
}

// ---------------------------------------------------------------------------

extern "C" void kernel_launch(void* const* d_in, const int* in_sizes, int n_in,
                              void* d_out, int out_size, void* d_ws, size_t ws_size,
                              hipStream_t stream)
{
    (void)in_sizes; (void)n_in; (void)out_size; (void)ws_size;

    const float* inputs     = (const float*)d_in[0];
    const float* doms       = (const float*)d_in[1];
    const int*   targets    = (const int*)d_in[2];
    const float* trans_W    = (const float*)d_in[3];
    const float* trans_b    = (const float*)d_in[4];
    const float* attnT_W    = (const float*)d_in[5];
    const float* attnT_b    = (const float*)d_in[6];
    const float* map_W      = (const float*)d_in[7];
    const float* map_b      = (const float*)d_in[8];
    const float* transO_W   = (const float*)d_in[9];
    const float* transO_b   = (const float*)d_in[10];
    const float* attnTO_W   = (const float*)d_in[11];
    const float* attnTO_b   = (const float*)d_in[12];
    const float* out_table  = (const float*)d_in[13];
    const float* Wih_f      = (const float*)d_in[14];
    const float* Whh_f      = (const float*)d_in[15];
    const float* b_f        = (const float*)d_in[16];
    const float* Wih_b      = (const float*)d_in[17];
    const float* Whh_b      = (const float*)d_in[18];
    const float* b_b        = (const float*)d_in[19];

    char* ws = (char*)d_ws;
    size_t off = 0;
    auto alloc = [&](size_t bytes) -> char* {
        off = (off + 255) & ~(size_t)255;
        char* p = ws + off;
        off += bytes;
        return p;
    };

    // fp32 domain-linear outputs
    float* t_f   = (float*)alloc(16L * 90000 * 4);
    float* at_f  = (float*)alloc(16L * 19329 * 4);
    float* to_f  = (float*)alloc(16L * 10000 * 4);
    float* ato_f = (float*)alloc(16L * 6529 * 4);
    // swizzled bf16 operands
    unsigned short* in_bf   = (unsigned short*)alloc(4096L * 320 * 2);   // A, Kp=320
    unsigned short* t_bf    = (unsigned short*)alloc(16L * 320 * 320 * 2);
    unsigned short* m1_bf   = (unsigned short*)alloc(16L * 320 * 64 * 2);
    unsigned short* WihT_fb = (unsigned short*)alloc(320L * 2048 * 2);
    unsigned short* WihT_bb = (unsigned short*)alloc(320L * 2048 * 2);
    unsigned short* WhhT_fb = (unsigned short*)alloc(512L * 2048 * 2);
    unsigned short* WhhT_bb = (unsigned short*)alloc(512L * 2048 * 2);
    unsigned short* mapWT   = (unsigned short*)alloc(1024L * 128 * 2);
    unsigned short* to_bf   = (unsigned short*)alloc(16L * 128 * 128 * 2);
    unsigned short* m1o_bf  = (unsigned short*)alloc(16L * 128 * 64 * 2);
    unsigned short* tab_bf  = (unsigned short*)alloc(128L * 32000 * 2);
    // activations
    float* trans_in = (float*)alloc(4096L * 300 * 4);
    float* hattn    = (float*)alloc(4096L * 64 * 4);
    unsigned short* w_bf = (unsigned short*)alloc(4096L * 320 * 2);
    float* pre_f    = (float*)alloc(4096L * 2048 * 4);
    float* pre_b    = (float*)alloc(4096L * 2048 * 4);
    float* hf       = (float*)alloc(4096L * 512 * 4);
    float* hb       = (float*)alloc(4096L * 512 * 4);
    unsigned short* cat_bf = (unsigned short*)alloc(4096L * 1024 * 2);
    float* proc     = (float*)alloc(4096L * 100 * 4);
    unsigned short* proc_bf = (unsigned short*)alloc(4096L * 128 * 2);
    float* trans_pr = (float*)alloc(4096L * 100 * 4);
    float* h2       = (float*)alloc(4096L * 64 * 4);
    unsigned short* pa_bf = (unsigned short*)alloc(4096L * 128 * 2);

    auto cdiv = [](long a, long b) -> unsigned { return (unsigned)((a + b - 1) / b); };

    // 1) domain-conditioned linears (fp32)
    dom_linear<<<cdiv(90000, 256), 256, 0, stream>>>(doms, trans_W,  trans_b,  t_f,   90000);
    dom_linear<<<cdiv(19329, 256), 256, 0, stream>>>(doms, attnT_W,  attnT_b,  at_f,  19329);
    dom_linear<<<cdiv(10000, 256), 256, 0, stream>>>(doms, transO_W, transO_b, to_f,  10000);
    dom_linear<<<cdiv(6529,  256), 256, 0, stream>>>(doms, attnTO_W, attnTO_b, ato_f, 6529);

    // 2) operand prep into swizzled bf16
    swizzleA_f32<<<dim3(cdiv(4096L * 320, 256), 1, 1), 256, 0, stream>>>(
        inputs, 0, 4096, 300, in_bf, 0, 4096, 320);
    swizzleB_f32<<<dim3(cdiv(320L * 320, 256), 1, 16), 256, 0, stream>>>(
        t_f, 90000, 300, 300, t_bf, 320L * 320, 320, 320);
    swizzleB_f32<<<dim3(cdiv(320L * 64, 256), 1, 16), 256, 0, stream>>>(
        at_f, 19329, 300, 64, m1_bf, 320L * 64, 320, 64);
    swizzleB_T<<<cdiv(320L * 2048, 256), 256, 0, stream>>>(Wih_f, 2048, 300, WihT_fb, 320, 2048);
    swizzleB_T<<<cdiv(320L * 2048, 256), 256, 0, stream>>>(Wih_b, 2048, 300, WihT_bb, 320, 2048);
    swizzleB_T<<<cdiv(512L * 2048, 256), 256, 0, stream>>>(Whh_f, 2048, 512, WhhT_fb, 512, 2048);
    swizzleB_T<<<cdiv(512L * 2048, 256), 256, 0, stream>>>(Whh_b, 2048, 512, WhhT_bb, 512, 2048);
    swizzleB_T<<<cdiv(1024L * 128, 256), 256, 0, stream>>>(map_W, 100, 1024, mapWT, 1024, 128);
    swizzleB_f32<<<dim3(cdiv(128L * 128, 256), 1, 16), 256, 0, stream>>>(
        to_f, 10000, 100, 100, to_bf, 128L * 128, 128, 128);
    swizzleB_f32<<<dim3(cdiv(128L * 64, 256), 1, 16), 256, 0, stream>>>(
        ato_f, 6529, 100, 64, m1o_bf, 128L * 64, 128, 64);
    swizzleB_f32<<<dim3(cdiv(128L * 32000, 256), 1, 1), 256, 0, stream>>>(
        out_table, 0, 100, 32000, tab_bf, 0, 128, 32000);

    // 3) trans_inputs[b] = inputs[b] @ t[b]   (batched; N=300, Np=320, KT=10)
    wmma_gemm<0><<<dim3(5, 16, 16), 128, 0, stream>>>(
        in_bf, 256L * 320, t_bf, 320L * 320, nullptr, 0,
        trans_in, 300, 256L * 300, 300, 10);

    // 4) attention hidden: leaky_relu(inputs @ m1 + b1)
    wmma_gemm<1><<<dim3(1, 16, 16), 128, 0, stream>>>(
        in_bf, 256L * 320, m1_bf, 320L * 64, at_f + 19200, 19329,
        hattn, 64, 256L * 64, 64, 10);

    // 5) aw + blend -> weighted (swizzled A, Kp=320)
    attn_blend1<<<4096, 320, 0, stream>>>(hattn, at_f, inputs, trans_in, w_bf);

    // 6) LSTM gate pre-GEMMs: [4096,320] @ [320,2048]
    wmma_gemm<0><<<dim3(32, 256, 1), 128, 0, stream>>>(
        w_bf, 0, WihT_fb, 0, nullptr, 0, pre_f, 2048, 0, 2048, 10);
    wmma_gemm<0><<<dim3(32, 256, 1), 128, 0, stream>>>(
        w_bf, 0, WihT_bb, 0, nullptr, 0, pre_b, 2048, 0, 2048, 10);

    // 7) recurrent biLSTM (persistent, one WGP per direction)
    lstm_kernel<<<dim3(2, 1, 1), 1024, 0, stream>>>(
        pre_f, pre_b, WhhT_fb, WhhT_bb, b_f, b_b, hf, hb);

    // 8) concat(h_f, h_b) -> swizzled A [4096,1024]
    swizzleA_cols<<<cdiv(4096L * 512, 256), 256, 0, stream>>>(hf, 512, cat_bf, 1024, 0,   4096L * 512);
    swizzleA_cols<<<cdiv(4096L * 512, 256), 256, 0, stream>>>(hb, 512, cat_bf, 1024, 512, 4096L * 512);

    // 9) proc = cat @ map_W.T + map_b   (N=100, Np=128, KT=32)
    wmma_gemm<0><<<dim3(2, 256, 1), 128, 0, stream>>>(
        cat_bf, 0, mapWT, 0, map_b, 0, proc, 100, 0, 100, 32);

    // 10) proc -> swizzled A [4096, 128]
    swizzleA_f32<<<dim3(cdiv(4096L * 128, 256), 1, 1), 256, 0, stream>>>(
        proc, 0, 4096, 100, proc_bf, 0, 4096, 128);

    // 11) trans_proc[b] = proc[b] @ t_out[b]   (KT=4)
    wmma_gemm<0><<<dim3(2, 16, 16), 128, 0, stream>>>(
        proc_bf, 256L * 128, to_bf, 128L * 128, nullptr, 0,
        trans_pr, 100, 256L * 100, 100, 4);

    // 12) output attention hidden
    wmma_gemm<1><<<dim3(1, 16, 16), 128, 0, stream>>>(
        proc_bf, 256L * 128, m1o_bf, 128L * 64, ato_f + 6400, 6529,
        h2, 64, 256L * 64, 64, 4);

    // 13) aw2 + blend -> proc_attn (swizzled A, Kp=128)
    attn_blend2<<<4096, 128, 0, stream>>>(h2, ato_f, proc, trans_pr, pa_bf);

    // 14) vocab projection: [4096,128] @ [128,32000] -> d_out (fp32)
    wmma_gemm<0><<<dim3(500, 256, 1), 128, 0, stream>>>(
        pa_bf, 0, tab_bf, 0, nullptr, 0,
        (float*)d_out, 32000, 0, 32000, 4);

    // 15) targets appended after the 4096*32000 logits
    tgt_copy<<<16, 256, 0, stream>>>(targets, (int*)d_out + 4096L * 32000, 4096);
}
